// PatchcoreModel_38242388803812
// MI455X (gfx1250) — matrix operand order repaired and checked by reading, not compiled, not run on hardware
//
#include <hip/hip_runtime.h>
#include <hip/hip_bf16.h>
#include <math.h>

// ---------------- problem constants ----------------
#define BATCH   16
#define C1      512
#define H1      28
#define W1      28
#define C2      1024
#define H2      14
#define W2      14
#define DIM     1536
#define PPI     784                 // patches per image (28*28)
#define NROWS   (BATCH * PPI)       // 12544 embedding rows
#define MBANK   15000
#define MPAD    15008               // padded to multiple of 16 for WMMA tiles
#define MSTEPS  (MPAD / 16)         // 938 column tiles
#define IMG     224
#define NNB     9
#define KS      33                  // gaussian taps, sigma = 4

// GEMM tiling
#define ROWS_PER_BLK 64             // 4 waves * 16 rows
#define KNN_THREADS  128
#define LDR          1544           // padded LDS row stride (bf16 elems): 3088B -> 4-bank rotate
#define A_ELEMS      (ROWS_PER_BLK * LDR)       // 98816
#define B_ELEMS      (16 * LDR)                 // 24704 per buffer
#define KNN_SHMEM    ((A_ELEMS + 2 * B_ELEMS) * 2)   // 296448 B <= 320KB WGP LDS

typedef __attribute__((ext_vector_type(8)))  float   v8f;
typedef __attribute__((ext_vector_type(8)))  __bf16  v8bf;
typedef __attribute__((ext_vector_type(16))) __bf16  v16bf;

union BfFrag { v16bf v; v8bf h[2]; };

__device__ __forceinline__ unsigned short f2bf(float f) {
    union { float f; unsigned u; } x; x.f = f;
    unsigned r = x.u + 0x7fffu + ((x.u >> 16) & 1u);   // round-to-nearest-even
    return (unsigned short)(r >> 16);
}
__device__ __forceinline__ unsigned pack2(float a, float b) {
    return (unsigned)f2bf(a) | ((unsigned)f2bf(b) << 16);
}

// ---------------- kernel 1: hierarchical embedding ----------------
__device__ __forceinline__ float pool3(const float* __restrict__ ch, int H, int W, int y, int x) {
    float s = 0.f;
    for (int dy = -1; dy <= 1; ++dy) {
        int yy = y + dy; if (yy < 0 || yy >= H) continue;
        for (int dx = -1; dx <= 1; ++dx) {
            int xx = x + dx; if (xx < 0 || xx >= W) continue;
            s += ch[yy * W + xx];
        }
    }
    return s * (1.0f / 9.0f);
}

__global__ void embed_kernel(const float* __restrict__ f1, const float* __restrict__ f2,
                             float* __restrict__ e) {
    int gid = blockIdx.x * 256 + threadIdx.x;
    if (gid >= NROWS * DIM) return;
    int c = gid % DIM;
    int p = (gid / DIM) % PPI;
    int b = gid / (DIM * PPI);
    int y = p / W1, x = p % W1;
    float val;
    if (c < C1) {
        val = pool3(f1 + ((size_t)(b * C1 + c)) * (H1 * W1), H1, W1, y, x);
    } else {
        int c2 = c - C1;
        const float* base = f2 + ((size_t)(b * C2 + c2)) * (H2 * W2);
        float sy = y * 0.5f - 0.25f, sx = x * 0.5f - 0.25f;
        int y0 = (int)floorf(sy); float fy = sy - (float)y0;
        int x0 = (int)floorf(sx); float fx = sx - (float)x0;
        int iy0 = min(max(y0, 0), H2 - 1), iy1 = min(max(y0 + 1, 0), H2 - 1);
        int ix0 = min(max(x0, 0), W2 - 1), ix1 = min(max(x0 + 1, 0), W2 - 1);
        float p00 = pool3(base, H2, W2, iy0, ix0);
        float p01 = pool3(base, H2, W2, iy0, ix1);
        float p10 = pool3(base, H2, W2, iy1, ix0);
        float p11 = pool3(base, H2, W2, iy1, ix1);
        val = (1.f - fy) * ((1.f - fx) * p00 + fx * p01)
            +        fy  * ((1.f - fx) * p10 + fx * p11);
    }
    e[gid] = val;   // gid == (b*PPI + p)*DIM + c
}

// ---------------- kernel 2: squared row norms (one wave per row) ----------------
__global__ void rownorm_kernel(const float* __restrict__ mat, float* __restrict__ norms,
                               int nrows, int nvalid) {
    int w = (blockIdx.x * blockDim.x + threadIdx.x) >> 5;
    int lane = threadIdx.x & 31;
    if (w >= nrows) return;
    float s;
    if (w < nvalid) {
        const float* r = mat + (size_t)w * DIM;
        s = 0.f;
        for (int k = lane; k < DIM; k += 32) { float v = r[k]; s += v * v; }
        for (int m = 16; m >= 1; m >>= 1) s += __shfl_xor(s, m, 32);
    } else {
        s = 1.0e30f;    // padded bank rows can never be nearest
    }
    if (lane == 0) norms[w] = s;
}

// ---------------- kernel 3: fused WMMA GEMM + min/argmin over memory bank ----------------
// Loads one 16-col B panel (fp32 bank rows -> bf16, conflict-free padded LDS) per thread.
__device__ __forceinline__ void load_bpanel(const float* __restrict__ bank,
                                            unsigned short* __restrict__ dstbase,
                                            int m0, int tid) {
    int c  = tid >> 3;                 // 0..15 panel column (bank row)
    int k0 = (tid & 7) * (DIM / 8);    // 192-elem segment
    int m  = m0 + c;
    unsigned short* dst = dstbase + c * LDR + k0;
    if (m < MBANK) {
        const float* src = bank + (size_t)m * DIM + k0;
#pragma unroll 4
        for (int k = 0; k < DIM / 8; k += 4) {
            float4 f = *(const float4*)(src + k);
            *(unsigned*)(dst + k)     = pack2(f.x, f.y);
            *(unsigned*)(dst + k + 2) = pack2(f.z, f.w);
        }
        int mn = m + 16;               // prefetch next panel (global_prefetch_b8)
        if (mn < MBANK) __builtin_prefetch(bank + (size_t)mn * DIM + k0, 0, 0);
    } else {
        for (int k = 0; k < DIM / 8; k += 2)
            *(unsigned*)(dst + k) = 0u;
    }
}

__device__ __forceinline__ void load_frags(BfFrag& a, BfFrag& b,
                                           const unsigned short* arow,
                                           const unsigned short* bcol,
                                           int kc, int khalf) {
    // A 16x32 bf16 frag: lane holds row lane16; K = 16*(j>>3) + 8*khalf + (j&7)
    a.h[0] = *(const v8bf*)(arow + kc + 8 * khalf);
    a.h[1] = *(const v8bf*)(arow + kc + 16 + 8 * khalf);
    // B 32x16 bf16 frag: lane holds col lane16; K = 16*(j>>3) + 8*khalf + (j&7)
    b.h[0] = *(const v8bf*)(bcol + kc + 8 * khalf);
    b.h[1] = *(const v8bf*)(bcol + kc + 16 + 8 * khalf);
}

// minWavesPerEU = 1: this kernel is LDS-bound to 1 block/WGP (296KB of 320KB),
// so tell the scheduler a single wave per SIMD is expected -> large VGPR budget
// -> it can rotate fragment register sets and keep >=8 ds_loads in flight
// instead of draining DScnt to zero before every WMMA.
__global__ __launch_bounds__(KNN_THREADS, 1)
void knn_gemm_kernel(const float* __restrict__ e, const float* __restrict__ bank,
                     const float* __restrict__ xnorm, const float* __restrict__ ynorm,
                     float* __restrict__ ps, int* __restrict__ loc) {
    extern __shared__ char smem[];
    unsigned short* As = (unsigned short*)smem;                 // [64][LDR] bf16
    unsigned short* Bs = As + A_ELEMS;                          // 2 x [16][LDR] bf16

    const int tid    = threadIdx.x;
    const int wave   = tid >> 5;
    const int lane   = tid & 31;
    const int lane16 = lane & 15;
    const int khalf  = lane >> 4;
    const int rowbase = blockIdx.x * ROWS_PER_BLK;

    // ---- load A block (64 rows x 1536) fp32 -> bf16 into padded LDS; 2 threads/row ----
    {
        int r = tid >> 1;
        int h = tid & 1;
        const float* src = e + (size_t)(rowbase + r) * DIM + h * (DIM / 2);
        unsigned short* dst = As + r * LDR + h * (DIM / 2);
#pragma unroll 4
        for (int k = 0; k < DIM / 2; k += 4) {
            float4 f = *(const float4*)(src + k);
            *(unsigned*)(dst + k)     = pack2(f.x, f.y);
            *(unsigned*)(dst + k + 2) = pack2(f.z, f.w);
        }
    }

    // per-lane row norms for this wave's 16 rows (C/D layout: row = j + 8*khalf)
    float xn[8];
#pragma unroll
    for (int j = 0; j < 8; ++j) xn[j] = xnorm[rowbase + wave * 16 + j + 8 * khalf];

    float rmin[8]; int ridx[8];
#pragma unroll
    for (int j = 0; j < 8; ++j) { rmin[j] = 3.0e38f; ridx[j] = 0; }

    const unsigned short* arow = As + (wave * 16 + lane16) * LDR;

    // preload panel 0, then double-buffer: issue panel i+1's global loads before
    // computing panel i so L2 streaming overlaps the 48 WMMAs.
    load_bpanel(bank, Bs, 0, tid);
    __syncthreads();

    for (int step = 0; step < MSTEPS; ++step) {
        const int m0  = step * 16;
        const int cur = step & 1;
        if (step + 1 < MSTEPS)
            load_bpanel(bank, Bs + (cur ^ 1) * B_ELEMS, m0 + 16, tid);

        const unsigned short* bcol = Bs + cur * B_ELEMS + lane16 * LDR;

        // ---- K loop: two independent accumulation chains, 2 fragment groups ahead.
        // Next-group loads target fresh registers and issue BEFORE this group's
        // WMMAs, so the waitcnt pass can use partial s_wait_dscnt and there is no
        // WMMA-source WAR hazard.
        v8f acc0 = {0.f, 0.f, 0.f, 0.f, 0.f, 0.f, 0.f, 0.f};
        v8f acc1 = {0.f, 0.f, 0.f, 0.f, 0.f, 0.f, 0.f, 0.f};
        BfFrag a0, b0, a1, b1, a0n, b0n, a1n, b1n;
        load_frags(a0, b0, arow, bcol, 0,  khalf);
        load_frags(a1, b1, arow, bcol, 32, khalf);
#pragma unroll
        for (int kc = 0; kc < DIM; kc += 64) {
            const bool more = (kc + 64 < DIM);
            if (more) {
                load_frags(a0n, b0n, arow, bcol, kc + 64, khalf);
                load_frags(a1n, b1n, arow, bcol, kc + 96, khalf);
            }
            acc0 = __builtin_amdgcn_wmma_f32_16x16x32_bf16(
                       false, a0.v, false, b0.v, (short)0, acc0, false, false);
            acc1 = __builtin_amdgcn_wmma_f32_16x16x32_bf16(
                       false, a1.v, false, b1.v, (short)0, acc1, false, false);
            if (more) { a0 = a0n; b0 = b0n; a1 = a1n; b1 = b1n; }
        }

        // ---- epilogue: per-lane running min/argmin (no shuffles in the hot loop) ----
        float yn = ynorm[m0 + lane16];
        int   ci = m0 + lane16;
#pragma unroll
        for (int j = 0; j < 8; ++j) {
            float v = xn[j] - 2.0f * (acc0[j] + acc1[j]) + yn;
            if (v < rmin[j]) { rmin[j] = v; ridx[j] = ci; }
        }
        __syncthreads();   // buf[cur] fully read; buf[cur^1] fully written
    }

    // ---- one final butterfly across the 16-lane column group ----
#pragma unroll
    for (int j = 0; j < 8; ++j) {
        float v  = rmin[j];
        int   ci = ridx[j];
#pragma unroll
        for (int s = 1; s < 16; s <<= 1) {
            float ov = __shfl_xor(v, s, 32);
            int   oc = __shfl_xor(ci, s, 32);
            if (ov < v || (ov == v && oc < ci)) { v = ov; ci = oc; }
        }
        if (lane16 == 0) {
            int row = rowbase + wave * 16 + j + 8 * khalf;
            ps[row]  = sqrtf(v > 0.f ? v : 0.f);
            loc[row] = ci;
        }
    }
}

// ---------------- kernel 4: per-image argmax of patch scores ----------------
__global__ void argmax_kernel(const float* __restrict__ ps, const int* __restrict__ loc,
                              float* __restrict__ score, int* __restrict__ nnidx,
                              int* __restrict__ mpbuf) {
    __shared__ float sv[256];
    __shared__ int   si[256];
    int b = blockIdx.x, tid = threadIdx.x;
    float best = -3.0e38f; int bi = 0;
    for (int p = tid; p < PPI; p += 256) {
        float v = ps[b * PPI + p];
        if (v > best || (v == best && p < bi)) { best = v; bi = p; }
    }
    sv[tid] = best; si[tid] = bi; __syncthreads();
    for (int s = 128; s > 0; s >>= 1) {
        if (tid < s) {
            if (sv[tid + s] > sv[tid] || (sv[tid + s] == sv[tid] && si[tid + s] < si[tid])) {
                sv[tid] = sv[tid + s]; si[tid] = si[tid + s];
            }
        }
        __syncthreads();
    }
    if (tid == 0) {
        int mp = si[0];
        mpbuf[b] = mp;
        score[b] = sv[0];
        nnidx[b] = loc[b * PPI + mp];
    }
}

// ---------------- kernel 5: distances of nn_sample to whole bank (one wave per (b,m)) ----------------
__global__ void nbdist_kernel(const float* __restrict__ bank, const int* __restrict__ nnidx,
                              float* __restrict__ dnb) {
    int gw = (blockIdx.x * blockDim.x + threadIdx.x) >> 5;
    int lane = threadIdx.x & 31;
    int b = gw / MPAD;
    int m = gw % MPAD;
    if (b >= BATCH) return;
    float out;
    if (m >= MBANK) {
        out = 1.0e30f;
    } else {
        const float* a = bank + (size_t)nnidx[b] * DIM;
        const float* c = bank + (size_t)m * DIM;
        float s = 0.f;
        for (int k = lane; k < DIM; k += 32) { float d = a[k] - c[k]; s += d * d; }
        for (int mm = 16; mm >= 1; mm >>= 1) s += __shfl_xor(s, mm, 32);
        out = sqrtf(fmaxf(s, 0.f));
    }
    if (lane == 0) dnb[b * MPAD + m] = out;
}

// ---------------- kernel 6: 9-NN selection + softmax re-weighting ----------------
__global__ void rescore_kernel(float* __restrict__ dnb, const float* __restrict__ e,
                               const float* __restrict__ bank, const float* __restrict__ score,
                               const int* __restrict__ mpbuf, float* __restrict__ pred) {
    __shared__ float sv[256];
    __shared__ int   si[256];
    __shared__ int   sup[NNB];
    __shared__ float d2[NNB];
    int b = blockIdx.x, tid = threadIdx.x;
    float* row = dnb + (size_t)b * MPAD;
    for (int s = 0; s < NNB; ++s) {
        float best = 3.0e38f; int bi = MPAD;
        for (int m = tid; m < MPAD; m += 256) {
            float v = row[m];
            if (v < best || (v == best && m < bi)) { best = v; bi = m; }
        }
        sv[tid] = best; si[tid] = bi; __syncthreads();
        for (int r = 128; r > 0; r >>= 1) {
            if (tid < r) {
                if (sv[tid + r] < sv[tid] || (sv[tid + r] == sv[tid] && si[tid + r] < si[tid])) {
                    sv[tid] = sv[tid + r]; si[tid] = si[tid + r];
                }
            }
            __syncthreads();
        }
        if (tid == 0) { sup[s] = si[0]; row[si[0]] = 3.0e38f; }
        __syncthreads();
    }
    if (tid < NNB) {
        const float* mf = e + (size_t)(b * PPI + mpbuf[b]) * DIM;
        const float* sf = bank + (size_t)sup[tid] * DIM;
        float s = 0.f;
        for (int k = 0; k < DIM; ++k) { float d = mf[k] - sf[k]; s += d * d; }
        d2[tid] = sqrtf(fmaxf(s, 0.f));
    }
    __syncthreads();
    if (tid == 0) {
        float mx = d2[0];
        for (int i = 1; i < NNB; ++i) mx = fmaxf(mx, d2[i]);
        float sum = 0.f;
        for (int i = 0; i < NNB; ++i) sum += __expf(d2[i] - mx);
        float w = 1.f - __expf(d2[0] - mx) / sum;
        pred[b] = w * score[b];
    }
}

// ---------------- kernel 7: 28->224 half-pixel bilinear upsample ----------------
__global__ void upsample_kernel(const float* __restrict__ ps, float* __restrict__ up) {
    int gid = blockIdx.x * 256 + threadIdx.x;
    if (gid >= BATCH * IMG * IMG) return;
    int x = gid % IMG;
    int y = (gid / IMG) % IMG;
    int b = gid / (IMG * IMG);
    float sy = y * 0.125f - 0.4375f, sx = x * 0.125f - 0.4375f;
    int y0 = (int)floorf(sy); float fy = sy - (float)y0;
    int x0 = (int)floorf(sx); float fx = sx - (float)x0;
    int iy0 = min(max(y0, 0), H1 - 1), iy1 = min(max(y0 + 1, 0), H1 - 1);
    int ix0 = min(max(x0, 0), W1 - 1), ix1 = min(max(x0 + 1, 0), W1 - 1);
    const float* p = ps + b * PPI;
    float v = (1.f - fy) * ((1.f - fx) * p[iy0 * W1 + ix0] + fx * p[iy0 * W1 + ix1])
            +        fy  * ((1.f - fx) * p[iy1 * W1 + ix0] + fx * p[iy1 * W1 + ix1]);
    up[gid] = v;
}

// ---------------- kernels 8/9: separable 33-tap gaussian blur, reflect padding ----------------
__device__ __forceinline__ void gweights(float* w) {
    float s = 0.f;
#pragma unroll
    for (int t = 0; t < KS; ++t) {
        float d = (float)t - 16.f;
        w[t] = __expf(-d * d * (1.f / 32.f));   // 2*sigma^2 = 32
        s += w[t];
    }
    float inv = 1.f / s;
#pragma unroll
    for (int t = 0; t < KS; ++t) w[t] *= inv;
}

__global__ void blurh_kernel(const float* __restrict__ in, float* __restrict__ out) {
    int gid = blockIdx.x * 256 + threadIdx.x;
    if (gid >= BATCH * IMG * IMG) return;
    int x = gid % IMG;
    int y = (gid / IMG) % IMG;
    int b = gid / (IMG * IMG);
    float w[KS]; gweights(w);
    const float* row = in + (size_t)b * IMG * IMG + y * IMG;
    float s = 0.f;
#pragma unroll
    for (int t = 0; t < KS; ++t) {
        int xx = x + t - 16;
        if (xx < 0) xx = -xx; else if (xx > IMG - 1) xx = 2 * (IMG - 1) - xx;
        s += w[t] * row[xx];
    }
    out[gid] = s;
}

__global__ void blurv_kernel(const float* __restrict__ in, float* __restrict__ out) {
    int gid = blockIdx.x * 256 + threadIdx.x;
    if (gid >= BATCH * IMG * IMG) return;
    int x = gid % IMG;
    int y = (gid / IMG) % IMG;
    int b = gid / (IMG * IMG);
    float w[KS]; gweights(w);
    const float* img = in + (size_t)b * IMG * IMG;
    float s = 0.f;
#pragma unroll
    for (int t = 0; t < KS; ++t) {
        int yy = y + t - 16;
        if (yy < 0) yy = -yy; else if (yy > IMG - 1) yy = 2 * (IMG - 1) - yy;
        s += w[t] * img[yy * IMG + x];
    }
    out[gid] = s;
}

// ---------------- host-side orchestration ----------------
extern "C" void kernel_launch(void* const* d_in, const int* in_sizes, int n_in,
                              void* d_out, int out_size, void* d_ws, size_t ws_size,
                              hipStream_t stream) {
    (void)in_sizes; (void)n_in; (void)out_size; (void)ws_size;
    const float* feat1 = (const float*)d_in[0];
    const float* feat2 = (const float*)d_in[1];
    const float* bank  = (const float*)d_in[2];
    float* out = (float*)d_out;     // [16*224*224 amap][16 pred_score]

    float* ws = (float*)d_ws;
    size_t off = 0;
    float* E    = ws + off; off += (size_t)NROWS * DIM;      // fp32 embeddings
    float* XN   = ws + off; off += NROWS;                    // |x|^2
    float* YN   = ws + off; off += MPAD;                     // |y|^2 (+inf pad)
    float* PS   = ws + off; off += NROWS;                    // patch scores
    int*   LOC  = (int*)(ws + off); off += NROWS;            // argmin indices
    float* DNB  = ws + off; off += (size_t)BATCH * MPAD;     // nn-to-bank distances
    float* UP   = ws + off; off += (size_t)BATCH * IMG * IMG;// upsampled map
    float* TMP  = ws + off; off += (size_t)BATCH * IMG * IMG;// blur intermediate
    float* SCORE = ws + off; off += BATCH;
    int*   MP    = (int*)(ws + off); off += BATCH;
    int*   NN    = (int*)(ws + off); off += BATCH;

    // 1. embedding (pool + bilinear + concat)
    embed_kernel<<<(NROWS * DIM + 255) / 256, 256, 0, stream>>>(feat1, feat2, E);
    // 2. norms
    rownorm_kernel<<<(NROWS + 7) / 8, 256, 0, stream>>>(E, XN, NROWS, NROWS);
    rownorm_kernel<<<(MPAD + 7) / 8, 256, 0, stream>>>(bank, YN, MPAD, MBANK);
    // 3. fused WMMA 1-NN search
    knn_gemm_kernel<<<NROWS / ROWS_PER_BLK, KNN_THREADS, KNN_SHMEM, stream>>>(
        E, bank, XN, YN, PS, LOC);
    // 4. image-level score path
    argmax_kernel<<<BATCH, 256, 0, stream>>>(PS, LOC, SCORE, NN, MP);
    nbdist_kernel<<<(BATCH * MPAD) / 8, 256, 0, stream>>>(bank, NN, DNB);
    rescore_kernel<<<BATCH, 256, 0, stream>>>(DNB, E, bank, SCORE, MP, out + BATCH * IMG * IMG);
    // 5. anomaly map
    int npix = BATCH * IMG * IMG;
    upsample_kernel<<<(npix + 255) / 256, 256, 0, stream>>>(PS, UP);
    blurh_kernel<<<(npix + 255) / 256, 256, 0, stream>>>(UP, TMP);
    blurv_kernel<<<(npix + 255) / 256, 256, 0, stream>>>(TMP, out);
}
